// GraphAttentionLayer_8804682957287
// MI455X (gfx1250) — matrix operand order repaired and checked by reading, not compiled, not run on hardware
//
#include <hip/hip_runtime.h>
#include <hip/hip_bf16.h>

typedef __attribute__((ext_vector_type(16))) _Float16 v16h;
typedef __attribute__((ext_vector_type(8)))  float    v8f;

#define N_ROWS 8192
#define F_IN   512
#define F_OUT  64

// ---------------------------------------------------------------- zero scratch
__global__ __launch_bounds__(256) void zero_ws_kernel(float* __restrict__ ws) {
    const int i = blockIdx.x * 256 + threadIdx.x;
    if (i < F_IN + F_OUT) ws[i] = 0.0f;
}

// ------------------------------------------------- column sums of h (16.8 MB)
// 256 threads/block, thread t owns columns 2t, 2t+1 (contiguous float2 -> fully
// coalesced). Grid-stride over rows; one atomic pair per thread at the end.
__global__ __launch_bounds__(256) void colsum_kernel(const float* __restrict__ h,
                                                     float* __restrict__ colsum) {
    const int c = threadIdx.x * 2;
    float sx = 0.0f, sy = 0.0f;
    for (int row = blockIdx.x; row < N_ROWS; row += gridDim.x) {
        const float2 v = *(const float2*)(h + (size_t)row * F_IN + c);
        // speculative streaming prefetch of the next row this thread will touch
        __builtin_prefetch(h + (size_t)(row + gridDim.x) * F_IN + c, 0, 0);
        sx += v.x;
        sy += v.y;
    }
    atomicAdd(&colsum[c],     sx);
    atomicAdd(&colsum[c + 1], sy);
}

// --------------------------------------- result = relu( (colsum/N) @ W ) via WMMA
// One wave32. A-matrix: hbar replicated into all 16 rows (all D rows identical).
// B-matrix: 16-column tile of W (f32 -> f16). K=512 accumulated in f32.
__global__ __launch_bounds__(32) void gemv_wmma_kernel(const float* __restrict__ colsum,
                                                       const float* __restrict__ W,
                                                       float* __restrict__ result) {
    __shared__ _Float16 hbar[F_IN];
    const int lane = threadIdx.x;

    for (int i = lane; i < F_IN; i += 32)
        hbar[i] = (_Float16)(colsum[i] * (1.0f / (float)N_ROWS));
    __syncthreads();

    // ISA 7.12.2 16-bit layouts (wave32):
    // A 16x32: lanes 0-15 -> K {0..7,16..23}; lanes 16-31 -> K {8..15,24..31}
    // B 32x16: same K order as A: lanes 0-15 -> K {0..7,16..23}; lanes 16-31 -> K {8..15,24..31}
    const int n  = lane & 15;
    const int aO = (lane < 16) ? 0 : 8;
    const int bO = (lane < 16) ? 0 : 16;

#pragma unroll
    for (int ct = 0; ct < 4; ++ct) {            // four 16-column tiles of W
        const int c0 = ct * 16;
        v8f acc = {};
#pragma unroll
        for (int kc = 0; kc < F_IN; kc += 32) { // 16 WMMA steps over K=512
            v16h A, B;
#pragma unroll
            for (int v = 0; v < 8; ++v) {
                const int ka = kc + aO + ((v < 4) ? (2 * v) : (16 + 2 * (v - 4)));
                A[2 * v]     = hbar[ka];
                A[2 * v + 1] = hbar[ka + 1];
                const int kb = kc + aO + ((v < 4) ? (2 * v) : (16 + 2 * (v - 4)));
                B[2 * v]     = (_Float16)W[(size_t)kb * F_OUT + c0 + n];
                B[2 * v + 1] = (_Float16)W[(size_t)(kb + 1) * F_OUT + c0 + n];
            }
            acc = __builtin_amdgcn_wmma_f32_16x16x32_f16(
                /*neg_a=*/false, A, /*neg_b=*/false, B,
                /*c_mod=*/(short)0, acc, /*reuse_a=*/false, /*reuse_b=*/false);
        }
        // D row 0 lives in acc[0] of lanes 0..15 (all rows identical anyway)
        if (lane < 16) result[c0 + lane] = fmaxf(acc[0], 0.0f);
    }
}

// ------------------------------------------------ broadcast 64 floats to 8192 rows
__global__ __launch_bounds__(256) void bcast_kernel(const float* __restrict__ result,
                                                    float4* __restrict__ out) {
    const int idx = blockIdx.x * 256 + threadIdx.x;   // 131072 float4 = 8192x64 f32
    const float4 r = ((const float4*)result)[idx & 15];
    out[idx] = r;
}

extern "C" void kernel_launch(void* const* d_in, const int* in_sizes, int n_in,
                              void* d_out, int out_size, void* d_ws, size_t ws_size,
                              hipStream_t stream) {
    // inputs: 0=h (8192x512 f32), 1=adj (unused!), 2=W (512x64 f32), 3=a (unused!)
    const float* h = (const float*)d_in[0];
    const float* W = (const float*)d_in[2];

    float* ws     = (float*)d_ws;
    float* colsum = ws;          // 512 floats
    float* result = ws + F_IN;   // 64 floats

    zero_ws_kernel  <<<3,   256, 0, stream>>>(ws);
    colsum_kernel   <<<512, 256, 0, stream>>>(h, colsum);
    gemv_wmma_kernel<<<1,    32, 0, stream>>>(colsum, W, result);
    bcast_kernel    <<<512, 256, 0, stream>>>(result, (float4*)d_out);
}
